// EGNNTimeDenoiser_33672543601318
// MI455X (gfx1250) — compile-verified, matrix operand order checked by reading, not run on hardware
//
#include <hip/hip_runtime.h>
#include <hip/hip_bf16.h>
#include <cstdint>

// ---------------------------------------------------------------------------
// EGNN time denoiser for MI455X (gfx1250), wave32 + WMMA f16 path.
// Edge MLP (E=524288, K=257->128->128) via v_wmma_f32_16x16x32_f16 with
// persistent blocks (weights resident in VGPRs) and async global->LDS
// gathers (GLOBAL_LOAD_ASYNC_TO_LDS_B128, ASYNCcnt-tracked).
// ---------------------------------------------------------------------------

typedef __attribute__((ext_vector_type(16))) _Float16 v16h;
typedef __attribute__((ext_vector_type(8)))  float    v8f;

namespace {
constexpr int kB      = 16;
constexpr int kL      = 1024;
constexpr int kN      = kB * kL;     // 16384 nodes
constexpr int kH      = 128;         // hidden dim
constexpr int kLayers = 4;
constexpr int kE      = 524288;      // edges
constexpr int kF      = 64;          // input feats
constexpr int kTilesPerBlock = 16;   // 16-edge tiles per persistent block
}

__device__ __forceinline__ float silu_f(float x) {
  return x / (1.0f + __expf(-x));
}

// Build a 16x32 f16 A-fragment (wave32 layout) from an LDS tile stored
// row-major as 16 rows x 128 halves. Lane L<16: M=L, K-halves {0..7,16..23};
// lane L>=16: M=L-16, K-halves {8..15,24..31} (pairs in increasing K).
__device__ __forceinline__ v16h load_a_frag(const _Float16* sm, int lane, int kc) {
  const int m  = lane & 15;
  const int kb = (lane < 16) ? 0 : 8;
  const _Float16* p = sm + m * 128 + kc * 32 + kb;
  union { v16h v; uint4 u[2]; } r;
  r.u[0] = *(const uint4*)(p);        // 8 halves, 16B aligned
  r.u[1] = *(const uint4*)(p + 16);   // 8 halves
  return r.v;
}

// Async copy of one 16-byte chunk from global into LDS (ASYNCcnt-tracked).
__device__ __forceinline__ void async_copy_b128(void* lds_dst, const void* gsrc) {
  unsigned lds_addr = (unsigned)(size_t)lds_dst;               // low 32b = LDS offset
  unsigned long long ga = (unsigned long long)(size_t)gsrc;
  asm volatile("global_load_async_to_lds_b128 %0, %1, off"
               :: "v"(lds_addr), "v"(ga) : "memory");
}

__device__ __forceinline__ void wait_asynccnt0() {
#if __has_builtin(__builtin_amdgcn_s_wait_asynccnt)
  __builtin_amdgcn_s_wait_asynccnt(0);
#else
  asm volatile("s_wait_asynccnt 0x0" ::: "memory");
#endif
}

// ---------------------------------------------------------------------------
// Weight prep: pack fp32 edge-MLP weights into per-lane fp16 B-fragments.
// Fragment f = kc*8+nt covers K rows [kc*32, kc*32+32), N cols [nt*16, nt*16+16).
// ---------------------------------------------------------------------------
__global__ void prep_w1_frags(const float* __restrict__ w1, _Float16* __restrict__ frag) {
  int idx = blockIdx.x * 256 + threadIdx.x;          // kLayers*64*512
  if (idx >= kLayers * 64 * 512) return;
  int i  = idx & 15;
  int ln = (idx >> 4) & 31;
  int f  = (idx >> 9) & 63;
  int l  = idx >> 15;
  int kc = f >> 3, nt = f & 7;
  int n  = ln & 15;
  int kb = (ln < 16) ? 0 : 8;
  int k  = kc * 32 + kb + ((i < 8) ? i : (8 + i));
  frag[idx] = (_Float16)w1[((size_t)l * 257 + k) * 128 + (nt * 16 + n)];
}

__global__ void prep_w2_frags(const float* __restrict__ w2, _Float16* __restrict__ frag) {
  int idx = blockIdx.x * 256 + threadIdx.x;          // kLayers*32*512
  if (idx >= kLayers * 32 * 512) return;
  int i  = idx & 15;
  int ln = (idx >> 4) & 31;
  int f  = (idx >> 9) & 31;
  int l  = idx >> 14;
  int kc = f >> 3, nt = f & 7;
  int n  = ln & 15;
  int kb = (ln < 16) ? 0 : 8;
  int k  = kc * 32 + kb + ((i < 8) ? i : (8 + i));
  frag[idx] = (_Float16)w2[((size_t)l * 128 + k) * 128 + (nt * 16 + n)];
}

// ---------------------------------------------------------------------------
// Embedding: h = [feats | x_t | t_norm] @ embed_w + embed_b  (fp32 + f16 mirror)
// ---------------------------------------------------------------------------
__global__ void embed_kernel(const float* __restrict__ feats,
                             const float* __restrict__ xt,
                             const int*   __restrict__ tarr,
                             const int*   __restrict__ Tptr,
                             const float* __restrict__ ew,
                             const float* __restrict__ eb,
                             float* __restrict__ h,
                             _Float16* __restrict__ hf16) {
  int idx = blockIdx.x * 256 + threadIdx.x;
  if (idx >= kN * kH) return;
  int n = idx >> 7, c = idx & 127;
  int b = n >> 10;
  float Tv = (float)Tptr[0];
  float tn = (float)tarr[b] / Tv;
  tn = fminf(fmaxf(tn, 0.0f), 1.0f);
  float acc = eb[c];
  const float* f = feats + (size_t)n * kF;
  #pragma unroll 8
  for (int k = 0; k < kF; ++k) acc += f[k] * ew[k * kH + c];
  acc += xt[n * 2 + 0] * ew[(kF + 0) * kH + c];
  acc += xt[n * 2 + 1] * ew[(kF + 1) * kH + c];
  acc += tn            * ew[(kF + 2) * kH + c];
  h[idx] = acc;
  hf16[idx] = (_Float16)acc;
}

// ---------------------------------------------------------------------------
// Edge MLP + scatter. Persistent block (8 waves) processes kTilesPerBlock
// 16-edge tiles; wave nt owns output columns [nt*16, nt*16+16) and keeps its
// 12 WMMA B-fragments in registers across all tiles.
// ---------------------------------------------------------------------------
__global__ __launch_bounds__(256)
void egnn_edge_kernel(const _Float16* __restrict__ hf16,
                      const float* __restrict__ pos,
                      const int* __restrict__ rowIdx,
                      const int* __restrict__ colIdx,
                      const _Float16* __restrict__ w1f,    // layer's 64 B-frags
                      const _Float16* __restrict__ w2f,    // layer's 32 B-frags
                      const float* __restrict__ w1last,    // edge_w1 row 256
                      const float* __restrict__ eb1,
                      const float* __restrict__ eb2,
                      const float* __restrict__ cw,        // coord_w[l]
                      const float* __restrict__ cb,        // coord_b[l]
                      float* __restrict__ agg,
                      float* __restrict__ posdelta) {
  __shared__ _Float16 s_h[2][16 * 128];   // gathered h[row], h[col] tiles
  __shared__ _Float16 s_m[16 * 128];      // edge-MLP hidden tile (transposed)
  __shared__ int   s_row[16], s_col[16];
  __shared__ float s_diff[16][3];
  __shared__ float s_d2[16];
  __shared__ float s_coef[16];

  const int tid   = threadIdx.x;
  const int lane  = tid & 31;
  const int nt    = tid >> 5;              // wave id == output n-tile
  const int ncol  = lane & 15;
  const int nglob = nt * 16 + ncol;
  const int mbase = (lane < 16) ? 0 : 8;

  // ---- Hoisted per-wave state: WMMA B-fragments + per-column scalars ------
  __builtin_prefetch(w1f + nt * 512 + lane * 16, 0, 0);   // global_prefetch_b8
  v16h w1b[8];
  #pragma unroll
  for (int kc = 0; kc < 8; ++kc)
    w1b[kc] = *(const v16h*)(w1f + (kc * 8 + nt) * 512 + lane * 16);
  v16h w2b[4];
  #pragma unroll
  for (int kc = 0; kc < 4; ++kc)
    w2b[kc] = *(const v16h*)(w2f + (kc * 8 + nt) * 512 + lane * 16);

  const float wl  = w1last[nglob];
  const float bb1 = eb1[nglob];
  const float bb2 = eb2[nglob];
  const float cwv = cw[nglob];
  const float cb0 = cb[0];

  for (int it = 0; it < kTilesPerBlock; ++it) {
    const int e0 = (blockIdx.x * kTilesPerBlock + it) * 16;

    if (tid < 16) {
      int r = rowIdx[e0 + tid];
      int c = colIdx[e0 + tid];
      s_row[tid] = r;
      s_col[tid] = c;
      float dx = pos[r * 3 + 0] - pos[c * 3 + 0];
      float dy = pos[r * 3 + 1] - pos[c * 3 + 1];
      float dz = pos[r * 3 + 2] - pos[c * 3 + 2];
      s_diff[tid][0] = dx; s_diff[tid][1] = dy; s_diff[tid][2] = dz;
      s_d2[tid]   = dx * dx + dy * dy + dz * dz;
      s_coef[tid] = 0.0f;
    }
    __syncthreads();

    // Async-gather the two 16x128 f16 tiles (512 x 16B chunks) into LDS.
    {
      #pragma unroll
      for (int t = 0; t < 2; ++t) {
        int c    = tid + t * 256;         // chunk id 0..511
        int tile = c >> 8;                // 0: h[row], 1: h[col]
        int m    = (c >> 4) & 15;
        int part = c & 15;                // 16B sub-chunk within 256B row
        int node = tile ? s_col[m] : s_row[m];
        const char* gsrc = (const char*)hf16 + (size_t)node * 256 + part * 16;
        async_copy_b128((char*)&s_h[0][0] + c * 16, gsrc);
      }
      wait_asynccnt0();
    }
    __syncthreads();

    // GEMM1: [h_row | h_col] (16x256) @ W1 (256x128), this wave's 16 cols.
    v8f acc = {};
    #pragma unroll
    for (int kc = 0; kc < 8; ++kc) {
      v16h a = load_a_frag((kc < 4) ? &s_h[0][0] : &s_h[1][0], lane, kc & 3);
      acc = __builtin_amdgcn_wmma_f32_16x16x32_f16(
          false, a, false, w1b[kc], (short)0, acc, false, false);
    }
    #pragma unroll
    for (int r = 0; r < 8; ++r) {
      int M = r + mbase;
      float v = acc[r] + s_d2[M] * wl + bb1;   // d2 rank-1 term + bias
      v = silu_f(v);
      s_m[M * 128 + nglob] = (_Float16)v;      // C-layout -> A-layout via LDS
    }
    __syncthreads();

    // GEMM2: m (16x128) @ W2 (128x128), this wave's 16 cols.
    v8f acc2 = {};
    #pragma unroll
    for (int kc = 0; kc < 4; ++kc) {
      v16h a = load_a_frag(s_m, lane, kc);
      acc2 = __builtin_amdgcn_wmma_f32_16x16x32_f16(
          false, a, false, w2b[kc], (short)0, acc2, false, false);
    }
    #pragma unroll
    for (int r = 0; r < 8; ++r) {
      int M = r + mbase;
      float v = silu_f(acc2[r] + bb2);
      atomicAdd(&agg[(size_t)s_row[M] * kH + nglob], v);  // segment_sum(m,row)
      atomicAdd(&s_coef[M], v * cwv);                     // m @ coord_w (LDS)
    }
    __syncthreads();

    if (tid < 16) {
      float cf = s_coef[tid] + cb0;
      int r = s_row[tid];
      atomicAdd(&posdelta[r * 3 + 0], s_diff[tid][0] * cf);
      atomicAdd(&posdelta[r * 3 + 1], s_diff[tid][1] * cf);
      atomicAdd(&posdelta[r * 3 + 2], s_diff[tid][2] * cf);
    }
    __syncthreads();
  }
}

// ---------------------------------------------------------------------------
// Node update, head, pos-update: bandwidth-trivial scalar kernels.
// ---------------------------------------------------------------------------
__global__ void node1_kernel(const float* __restrict__ h,
                             const float* __restrict__ agg,
                             const float* __restrict__ nw1,   // [256,128]
                             const float* __restrict__ nb1,
                             float* __restrict__ hid) {
  int idx = blockIdx.x * 256 + threadIdx.x;
  if (idx >= kN * kH) return;
  int n = idx >> 7, c = idx & 127;
  float acc = nb1[c];
  const float* hr = h   + (size_t)n * kH;
  const float* ar = agg + (size_t)n * kH;
  #pragma unroll 8
  for (int k = 0; k < kH; ++k) acc += hr[k] * nw1[k * kH + c];
  #pragma unroll 8
  for (int k = 0; k < kH; ++k) acc += ar[k] * nw1[(kH + k) * kH + c];
  hid[idx] = silu_f(acc);
}

__global__ void node2_kernel(const float* __restrict__ hid,
                             const float* __restrict__ nw2,   // [128,128]
                             const float* __restrict__ nb2,
                             float* __restrict__ h,
                             _Float16* __restrict__ hf16) {
  int idx = blockIdx.x * 256 + threadIdx.x;
  if (idx >= kN * kH) return;
  int n = idx >> 7, c = idx & 127;
  float acc = nb2[c];
  const float* hr = hid + (size_t)n * kH;
  #pragma unroll 8
  for (int k = 0; k < kH; ++k) acc += hr[k] * nw2[k * kH + c];
  float nh = h[idx] + acc;
  h[idx] = nh;
  hf16[idx] = (_Float16)nh;
}

__global__ void pos_update_kernel(float* __restrict__ pos,
                                  const float* __restrict__ delta) {
  int idx = blockIdx.x * 256 + threadIdx.x;
  if (idx >= kN * 3) return;
  pos[idx] += delta[idx];
}

__global__ void head1_kernel(const float* __restrict__ h,
                             const float* __restrict__ hw1,
                             const float* __restrict__ hb1,
                             float* __restrict__ hd) {
  int idx = blockIdx.x * 256 + threadIdx.x;
  if (idx >= kN * kH) return;
  int n = idx >> 7, c = idx & 127;
  float acc = hb1[c];
  const float* hr = h + (size_t)n * kH;
  #pragma unroll 8
  for (int k = 0; k < kH; ++k) acc += hr[k] * hw1[k * kH + c];
  hd[idx] = silu_f(acc);
}

__global__ void head2_kernel(const float* __restrict__ hd,
                             const float* __restrict__ hw2,   // [128,2]
                             const float* __restrict__ hb2,
                             const float* __restrict__ mask,
                             float* __restrict__ out) {
  int idx = blockIdx.x * 256 + threadIdx.x;
  if (idx >= kN * 2) return;
  int n = idx >> 1, j = idx & 1;
  float acc = hb2[j];
  const float* hr = hd + (size_t)n * kH;
  #pragma unroll 8
  for (int k = 0; k < kH; ++k) acc += hr[k] * hw2[k * 2 + j];
  out[idx] = acc * mask[n];
}

// ---------------------------------------------------------------------------
// Host-side orchestration (graph-capture safe: only async ops on `stream`).
// ---------------------------------------------------------------------------
extern "C" void kernel_launch(void* const* d_in, const int* in_sizes, int n_in,
                              void* d_out, int out_size, void* d_ws, size_t ws_size,
                              hipStream_t stream) {
  (void)in_sizes; (void)n_in; (void)out_size; (void)ws_size;

  const float* feats   = (const float*)d_in[0];
  const float* pos_in  = (const float*)d_in[1];
  const int*   eidx    = (const int*)d_in[2];     // [2,E]
  const float* x_t     = (const float*)d_in[3];
  const int*   t_arr   = (const int*)d_in[4];
  const int*   T_ptr   = (const int*)d_in[5];
  const float* mask    = (const float*)d_in[6];
  const float* embed_w = (const float*)d_in[7];
  const float* embed_b = (const float*)d_in[8];
  const float* edge_w1 = (const float*)d_in[9];   // [L,257,128]
  const float* edge_b1 = (const float*)d_in[10];  // [L,128]
  const float* edge_w2 = (const float*)d_in[11];  // [L,128,128]
  const float* edge_b2 = (const float*)d_in[12];  // [L,128]
  const float* coord_w = (const float*)d_in[13];  // [L,128,1]
  const float* coord_b = (const float*)d_in[14];  // [L,1]
  const float* node_w1 = (const float*)d_in[15];  // [L,256,128]
  const float* node_b1 = (const float*)d_in[16];
  const float* node_w2 = (const float*)d_in[17];  // [L,128,128]
  const float* node_b2 = (const float*)d_in[18];
  const float* head_w1 = (const float*)d_in[19];
  const float* head_b1 = (const float*)d_in[20];
  const float* head_w2 = (const float*)d_in[21];  // [128,2]
  const float* head_b2 = (const float*)d_in[22];
  float* out = (float*)d_out;

  // Workspace carve-up.
  char* ws = (char*)d_ws;
  size_t off = 0;
  float*    h_buf    = (float*)(ws + off);    off += (size_t)kN * kH * 4;   // 8 MB
  _Float16* hf16     = (_Float16*)(ws + off); off += (size_t)kN * kH * 2;   // 4 MB
  float*    agg      = (float*)(ws + off);    off += (size_t)kN * kH * 4;   // 8 MB
  float*    hid      = (float*)(ws + off);    off += (size_t)kN * kH * 4;   // 8 MB
  float*    pos_cur  = (float*)(ws + off);    off += (size_t)kN * 3 * 4;
  float*    pos_del  = (float*)(ws + off);    off += (size_t)kN * 3 * 4;
  _Float16* w1f      = (_Float16*)(ws + off); off += (size_t)kLayers * 64 * 512 * 2;
  _Float16* w2f      = (_Float16*)(ws + off); off += (size_t)kLayers * 32 * 512 * 2;

  const int* rowIdx = eidx;
  const int* colIdx = eidx + kE;

  // Mutable positions.
  hipMemcpyAsync(pos_cur, pos_in, (size_t)kN * 3 * 4,
                 hipMemcpyDeviceToDevice, stream);

  // Pack edge-MLP weights into WMMA B-fragment order (fp16).
  prep_w1_frags<<<(kLayers * 64 * 512 + 255) / 256, 256, 0, stream>>>(edge_w1, w1f);
  prep_w2_frags<<<(kLayers * 32 * 512 + 255) / 256, 256, 0, stream>>>(edge_w2, w2f);

  // Embedding.
  embed_kernel<<<(kN * kH + 255) / 256, 256, 0, stream>>>(
      feats, x_t, t_arr, T_ptr, embed_w, embed_b, h_buf, hf16);

  const int edgeBlocks = kE / (16 * kTilesPerBlock);   // 2048 persistent blocks

  for (int l = 0; l < kLayers; ++l) {
    hipMemsetAsync(agg, 0, (size_t)kN * kH * 4, stream);
    hipMemsetAsync(pos_del, 0, (size_t)kN * 3 * 4, stream);

    egnn_edge_kernel<<<edgeBlocks, 256, 0, stream>>>(
        hf16, pos_cur, rowIdx, colIdx,
        w1f + (size_t)l * 64 * 512,
        w2f + (size_t)l * 32 * 512,
        edge_w1 + ((size_t)l * 257 + 256) * 128,
        edge_b1 + (size_t)l * 128,
        edge_b2 + (size_t)l * 128,
        coord_w + (size_t)l * 128,
        coord_b + l,
        agg, pos_del);

    node1_kernel<<<(kN * kH + 255) / 256, 256, 0, stream>>>(
        h_buf, agg, node_w1 + (size_t)l * 256 * 128, node_b1 + (size_t)l * 128, hid);
    node2_kernel<<<(kN * kH + 255) / 256, 256, 0, stream>>>(
        hid, node_w2 + (size_t)l * 128 * 128, node_b2 + (size_t)l * 128, h_buf, hf16);
    pos_update_kernel<<<(kN * 3 + 255) / 256, 256, 0, stream>>>(pos_cur, pos_del);
  }

  head1_kernel<<<(kN * kH + 255) / 256, 256, 0, stream>>>(h_buf, head_w1, head_b1, hid);
  head2_kernel<<<(kN * 2 + 255) / 256, 256, 0, stream>>>(hid, head_w2, head_b2, mask, out);
}
